// DecoderTreeRNN_71038759076049
// MI455X (gfx1250) — matrix operand order repaired and checked by reading, not compiled, not run on hardware
//
#include <hip/hip_runtime.h>
#include <hip/hip_bf16.h>
#include <math.h>
#include <stdint.h>

typedef __attribute__((ext_vector_type(16))) _Float16     v16h;
typedef __attribute__((ext_vector_type(8)))  _Float16     v8h;
typedef __attribute__((ext_vector_type(8)))  float        v8f;
typedef __attribute__((ext_vector_type(4)))  unsigned int u32x4;
typedef __attribute__((ext_vector_type(8)))  int          i32x8;
typedef __attribute__((ext_vector_type(4)))  int          i32x4;

#define HID    1024
#define VOCAB  32000
#define BATCH  64
#define TDEPTH 5

// TDM availability (ROCm7.2 clang-22: 5-arg; amdgpu-toolchain clang-23: 6-arg)
#if defined(__has_builtin)
#if __has_builtin(__builtin_amdgcn_tensor_load_to_lds) && __has_builtin(__builtin_amdgcn_s_wait_tensorcnt)
#define USE_TDM 1
#endif
#endif
#ifndef USE_TDM
#define USE_TDM 0
#endif

// ---------------- fragment loaders (contiguous k per lane half, same for A and B) ----------------
// A 16x32 f16 frag, lane l: row M = l&15; k-base = (l<16)?0:16.
// elements 0..15 <- k = kb+0..15 contiguous
__device__ __forceinline__ v16h load_a_frag(const _Float16* p) {
  v8h lo = *(const v8h*)p;
  v8h hi = *(const v8h*)(p + 8);
  return __builtin_shufflevector(lo, hi, 0,1,2,3,4,5,6,7,8,9,10,11,12,13,14,15);
}
__device__ __forceinline__ v16h load_a_frag(const float* p) {
  v16h r;
#pragma unroll
  for (int i = 0; i < 8; ++i) { r[i] = (_Float16)p[i]; r[i + 8] = (_Float16)p[i + 8]; }
  return r;
}
// B 32x16 f16 frag, lane l: col N = l&15; k = ((l<16)?0:16) + 0..15 contiguous
__device__ __forceinline__ v16h load_b_frag(const _Float16* p) { return *(const v16h*)p; }
__device__ __forceinline__ v16h load_b_frag(const float* p) {
  v16h r;
#pragma unroll
  for (int i = 0; i < 16; ++i) r[i] = (_Float16)p[i];
  return r;
}

#define WMMA_F16(a, b, c) \
  __builtin_amdgcn_wmma_f32_16x16x32_f16(false, (a), false, (b), (short)0, (c), false, false)

// ---------------- simple per-wave 32x32 GEMM (fallback + small levels) ----------------
template <typename T>
__global__ void gemm_wmma(const T* __restrict__ A, const T* __restrict__ W,
                          const float* __restrict__ bias, float* __restrict__ C,
                          int R, int N, int K, int remapInner, int remapOuter) {
  const int lane = threadIdx.x & 31;
  const int wave = threadIdx.x >> 5;
  const int nTilesN = N >> 5;
  const int nTilesM = R >> 5;
  const int tile = blockIdx.x * (blockDim.x >> 5) + wave;
  if (tile >= nTilesM * nTilesN) return;
  const int tm = tile / nTilesN;
  const int tn = tile - tm * nTilesN;
  const int l16 = lane & 15;
  const int hi  = lane >> 4;

  const T* Ar0 = A + (size_t)(tm * 32 + l16) * K + (hi ? 16 : 0);
  const T* Ar1 = Ar0 + (size_t)16 * K;
  const T* Wr0 = W + (size_t)(tn * 32 + l16) * K + (hi ? 16 : 0);
  const T* Wr1 = Wr0 + (size_t)16 * K;

  v8f acc00 = {}, acc01 = {}, acc10 = {}, acc11 = {};
  for (int kk = 0; kk < K; kk += 32) {
    v16h a0 = load_a_frag(Ar0 + kk);
    v16h a1 = load_a_frag(Ar1 + kk);
    v16h b0 = load_b_frag(Wr0 + kk);
    v16h b1 = load_b_frag(Wr1 + kk);
    acc00 = WMMA_F16(a0, b0, acc00);
    acc01 = WMMA_F16(a0, b1, acc01);
    acc10 = WMMA_F16(a1, b0, acc10);
    acc11 = WMMA_F16(a1, b1, acc11);
  }
  const int mb = tm * 32 + (hi ? 8 : 0);
  const int n0 = tn * 32 + l16;
  const int n1 = n0 + 16;
  const float bv0 = bias ? bias[n0] : 0.0f;
  const float bv1 = bias ? bias[n1] : 0.0f;
#pragma unroll
  for (int v = 0; v < 8; ++v) {
    int m0 = mb + v, m1 = mb + 16 + v;
    int o0 = remapInner ? (m0 % remapInner) * remapOuter + m0 / remapInner : m0;
    int o1 = remapInner ? (m1 % remapInner) * remapOuter + m1 / remapInner : m1;
    C[(size_t)o0 * N + n0] = acc00[v] + bv0;
    C[(size_t)o0 * N + n1] = acc01[v] + bv1;
    C[(size_t)o1 * N + n0] = acc10[v] + bv0;
    C[(size_t)o1 * N + n1] = acc11[v] + bv1;
  }
}

// ---------------- TDM descriptor (ISA ch.8 D#) ----------------
__device__ __forceinline__ unsigned lds_off_of(const void* p) {
  // addrspace(3) generic pointer: low 32 bits are the LDS byte offset (ISA 10.2)
  return (unsigned)(uintptr_t)p;
}

#if USE_TDM
// 2D tile load: tile0 (contig, elems) x tile1 (rows), 16-bit elems, row stride
// stride0 elems; LDS rows padded via TDM pad: every 16 DWORDs (64B) add 4 DWORDs
// (16B) -> 80B LDS pitch (40 halves), bank-conflict-free fragment reads.
__device__ __forceinline__ void tdm_load_2d(const void* gp, unsigned ldsOff,
                                            unsigned td0, unsigned td1,
                                            unsigned tile0, unsigned tile1,
                                            unsigned long long stride0) {
  unsigned long long ga = (unsigned long long)(uintptr_t)gp;
  u32x4 g0 = { 1u,                                   // count=1 (valid), user D#
               ldsOff,                               // lds_addr (bytes)
               (unsigned)ga,                         // global_addr[31:0]
               ((unsigned)(ga >> 32) & 0x01FFFFFFu) | (2u << 30) };  // addr[56:32]|type=2
  i32x8 g1;
  g1[0] = (int)((1u << 16) | (1u << 20) | (3u << 22) | (3u << 25)); // dsize=2B,pad on,16DW+4DW
  g1[1] = (int)((td0 & 0xFFFFu) << 16);                             // tensor_dim0 lo
  g1[2] = (int)(((td0 >> 16) & 0xFFFFu) | ((td1 & 0xFFFFu) << 16)); // td0 hi | td1 lo
  g1[3] = (int)(((td1 >> 16) & 0xFFFFu) | ((tile0 & 0xFFFFu) << 16)); // td1 hi | tile_dim0
  g1[4] = (int)(tile1 & 0xFFFFu);                                   // tile_dim1 (tile_dim2=0)
  g1[5] = (int)(unsigned)(stride0 & 0xFFFFFFFFu);                   // dim0_stride lo
  g1[6] = (int)((unsigned)((stride0 >> 32) & 0xFFFFu));             // dim0_stride hi
  g1[7] = 0;
  i32x4 z4 = { 0, 0, 0, 0 };
#if defined(__clang_major__) && (__clang_major__ >= 23)
  i32x8 z8 = { 0, 0, 0, 0, 0, 0, 0, 0 };
  __builtin_amdgcn_tensor_load_to_lds(g0, g1, z4, z4, z8, 0);
#else
  __builtin_amdgcn_tensor_load_to_lds(g0, g1, z4, z4, 0);
#endif
}
#endif

// ---------------- LDS-tiled 128x128 block GEMM, TDM double-buffered ----------------
#define BM 128
#define BN 128
#define BK 32
#define PITCH 40  // halves per LDS row (80B)

__device__ __forceinline__ void coop_load_tile(_Float16* dst, const _Float16* src, int K) {
  // 128 rows x 32 halves; 256 threads copy one 32B chunk each x2 rows-halves
  const int t   = threadIdx.x;
  const int row = t >> 1;
  const int co  = (t & 1) * 16;
  const _Float16* s = src + (size_t)row * K + co;
  _Float16* d = dst + row * PITCH + co;
  *(v8h*)d       = *(const v8h*)s;
  *(v8h*)(d + 8) = *(const v8h*)(s + 8);
}

__global__ void __launch_bounds__(256)
gemm_wmma_tiled(const _Float16* __restrict__ A, const _Float16* __restrict__ W,
                const float* __restrict__ bias, float* __restrict__ C,
                int R, int N, int K, int remapInner, int remapOuter) {
  __shared__ _Float16 sA[2][BM * PITCH];
  __shared__ _Float16 sB[2][BN * PITCH];

  const int lane = threadIdx.x & 31;
  const int wave = threadIdx.x >> 5;
  const int l16  = lane & 15;
  const int hi   = lane >> 4;
  const int bn   = N / BN;
  const int tm   = blockIdx.x / bn;
  const int tn   = blockIdx.x - tm * bn;
  const int wm   = (wave >> 2) * 64;  // wave grid 2(M) x 4(N)
  const int wn   = (wave & 3) * 32;

  const _Float16* Ag = A + (size_t)(tm * BM) * K;
  const _Float16* Wg = W + (size_t)(tn * BN) * K;
  const int nk = K / BK;

#if USE_TDM
  if (wave == 0) {
    tdm_load_2d(Ag, lds_off_of(&sA[0][0]), (unsigned)K, (unsigned)BM, BK, BM, (unsigned)K);
    tdm_load_2d(Wg, lds_off_of(&sB[0][0]), (unsigned)K, (unsigned)BN, BK, BN, (unsigned)K);
    __builtin_amdgcn_s_wait_tensorcnt(0);
  }
  __syncthreads();
#endif

  v8f acc[4][2] = {};
  for (int kt = 0; kt < nk; ++kt) {
#if USE_TDM
    const int buf = kt & 1;
    if (wave == 0 && kt + 1 < nk) {  // prefetch next K-slab into the other buffer
      tdm_load_2d(Ag + (kt + 1) * BK, lds_off_of(&sA[buf ^ 1][0]),
                  (unsigned)(K - (kt + 1) * BK), (unsigned)BM, BK, BM, (unsigned)K);
      tdm_load_2d(Wg + (kt + 1) * BK, lds_off_of(&sB[buf ^ 1][0]),
                  (unsigned)(K - (kt + 1) * BK), (unsigned)BN, BK, BN, (unsigned)K);
    }
#else
    const int buf = 0;
    coop_load_tile(&sA[0][0], Ag + kt * BK, K);
    coop_load_tile(&sB[0][0], Wg + kt * BK, K);
    __syncthreads();
#endif
    const _Float16* sAb = &sA[buf][0];
    const _Float16* sBb = &sB[buf][0];
    v16h b0 = *(const v16h*)(sBb + (wn + l16) * PITCH + (hi ? 16 : 0));
    v16h b1 = *(const v16h*)(sBb + (wn + 16 + l16) * PITCH + (hi ? 16 : 0));
    v16h a[4];
#pragma unroll
    for (int mf = 0; mf < 4; ++mf)
      a[mf] = load_a_frag(sAb + (wm + mf * 16 + l16) * PITCH + (hi ? 16 : 0));
#pragma unroll
    for (int mf = 0; mf < 4; ++mf) {
      acc[mf][0] = WMMA_F16(a[mf], b0, acc[mf][0]);
      acc[mf][1] = WMMA_F16(a[mf], b1, acc[mf][1]);
    }
#if USE_TDM
    if (wave == 0 && kt + 1 < nk) __builtin_amdgcn_s_wait_tensorcnt(0);
#endif
    __syncthreads();
  }

#pragma unroll
  for (int mf = 0; mf < 4; ++mf) {
    const int mb = tm * BM + wm + mf * 16 + (hi ? 8 : 0);
#pragma unroll
    for (int nf = 0; nf < 2; ++nf) {
      const int n = tn * BN + wn + nf * 16 + l16;
      const float bv = bias ? bias[n] : 0.0f;
#pragma unroll
      for (int v = 0; v < 8; ++v) {
        int m = mb + v;
        int o = remapInner ? (m % remapInner) * remapOuter + m / remapInner : m;
        C[(size_t)o * N + n] = acc[mf][nf][v] + bv;
      }
    }
  }
}

// ---------------- fp32 -> f16 conversion ----------------
__global__ void cvt_f32_f16(const float* __restrict__ s, _Float16* __restrict__ d, int n) {
  int i = blockIdx.x * blockDim.x + threadIdx.x;
  if (i < n) d[i] = (_Float16)s[i];
}

// ---------------- GRU gate combine + child interleave ----------------
__global__ void gru_gates(const float* __restrict__ h, const float* __restrict__ gl,
                          const float* __restrict__ gr, const float* __restrict__ bihl,
                          const float* __restrict__ bihr, float* __restrict__ hnext,
                          int P, int nodes) {
  int idx = blockIdx.x * blockDim.x + threadIdx.x;
  if (idx >= P * HID) return;
  int p = idx / HID;
  int j = idx - p * HID;
  int b = p / nodes;
  int node = p - b * nodes;
  float hv = h[idx];
  size_t gbase = (size_t)p * (3 * HID) + j;
  size_t obase = ((size_t)(b * (2 * nodes) + 2 * node)) * HID + j;
  {
    float r  = 1.0f / (1.0f + expf(-(gl[gbase] + bihl[j])));
    float z  = 1.0f / (1.0f + expf(-(gl[gbase + HID] + bihl[HID + j])));
    float nn = tanhf(bihl[2 * HID + j] + r * gl[gbase + 2 * HID]);
    hnext[obase] = (1.0f - z) * nn + z * hv;
  }
  {
    float r  = 1.0f / (1.0f + expf(-(gr[gbase] + bihr[j])));
    float z  = 1.0f / (1.0f + expf(-(gr[gbase + HID] + bihr[HID + j])));
    float nn = tanhf(bihr[2 * HID + j] + r * gr[gbase + 2 * HID]);
    hnext[obase + HID] = (1.0f - z) * nn + z * hv;
  }
}

// ---------------- in-place log-softmax, one 256-thread block per row ----------------
__global__ void log_softmax_inplace(float* __restrict__ x, int V) {
  float* row = x + (size_t)blockIdx.x * V;
  __shared__ float sred[8];
  const int t = threadIdx.x, lane = t & 31, w = t >> 5;
  float m = -INFINITY;
  for (int i = t; i < V; i += 256) m = fmaxf(m, row[i]);
#pragma unroll
  for (int o = 16; o > 0; o >>= 1) m = fmaxf(m, __shfl_xor(m, o, 32));
  if (lane == 0) sred[w] = m;
  __syncthreads();
  float rmax = sred[0];
#pragma unroll
  for (int i = 1; i < 8; ++i) rmax = fmaxf(rmax, sred[i]);
  __syncthreads();
  float s = 0.0f;
  for (int i = t; i < V; i += 256) s += expf(row[i] - rmax);
#pragma unroll
  for (int o = 16; o > 0; o >>= 1) s += __shfl_xor(s, o, 32);
  if (lane == 0) sred[w] = s;
  __syncthreads();
  float tot = 0.0f;
#pragma unroll
  for (int i = 0; i < 8; ++i) tot += sred[i];
  const float lse = rmax + logf(tot);
  for (int i = t; i < V; i += 256) row[i] -= lse;
}

// ---------------- host orchestration ----------------
extern "C" void kernel_launch(void* const* d_in, const int* in_sizes, int n_in,
                              void* d_out, int out_size, void* d_ws, size_t ws_size,
                              hipStream_t stream) {
  const float* enc  = (const float*)d_in[0];
  const float* WhhL = (const float*)d_in[1];
  const float* bihL = (const float*)d_in[2];
  const float* bhhL = (const float*)d_in[3];
  const float* WhhR = (const float*)d_in[4];
  const float* bihR = (const float*)d_in[5];
  const float* bhhR = (const float*)d_in[6];
  const float* Wout = (const float*)d_in[7];
  const float* bout = (const float*)d_in[8];
  float* out = (float*)d_out;

  const int LEAVES = 1 << TDEPTH;        // 32
  const int ROWS   = BATCH * LEAVES;     // 2048

  char* ws = (char*)d_ws;
  float* hA = (float*)ws;  ws += (size_t)ROWS * HID * 4;
  float* hB = (float*)ws;  ws += (size_t)ROWS * HID * 4;
  float* gL = (float*)ws;  ws += (size_t)(ROWS / 2) * 3 * HID * 4;
  float* gR = (float*)ws;  ws += (size_t)(ROWS / 2) * 3 * HID * 4;
  _Float16* hH   = (_Float16*)ws; ws += (size_t)ROWS * HID * 2;
  _Float16* WL16 = (_Float16*)ws; ws += (size_t)3 * HID * HID * 2;
  _Float16* WR16 = (_Float16*)ws; ws += (size_t)3 * HID * HID * 2;
  _Float16* WO16 = (_Float16*)ws; ws += (size_t)VOCAB * HID * 2;
  const size_t fullNeed = (size_t)(ws - (char*)d_ws);
  const bool pre = ws_size >= fullNeed;  // f16-weights fast path

  hipMemcpyAsync(hA, enc, (size_t)BATCH * HID * 4, hipMemcpyDeviceToDevice, stream);

  if (pre) {
    const int nw = 3 * HID * HID;
    cvt_f32_f16<<<(nw + 255) / 256, 256, 0, stream>>>(WhhL, WL16, nw);
    cvt_f32_f16<<<(nw + 255) / 256, 256, 0, stream>>>(WhhR, WR16, nw);
    const int no = VOCAB * HID;
    cvt_f32_f16<<<(no + 255) / 256, 256, 0, stream>>>(Wout, WO16, no);
  }

  for (int l = 0; l < TDEPTH; ++l) {
    const int P = BATCH << l;
    if (pre) {
      const int nh = P * HID;
      cvt_f32_f16<<<(nh + 255) / 256, 256, 0, stream>>>(hA, hH, nh);
      if (P % BM == 0) {
        const int blocks = (P / BM) * ((3 * HID) / BN);
        gemm_wmma_tiled<<<blocks, 256, 0, stream>>>(hH, WL16, bhhL, gL, P, 3 * HID, HID, 0, 0);
        gemm_wmma_tiled<<<blocks, 256, 0, stream>>>(hH, WR16, bhhR, gR, P, 3 * HID, HID, 0, 0);
      } else {
        const int blocks = ((P / 32) * ((3 * HID) / 32) + 7) / 8;
        gemm_wmma<_Float16><<<blocks, 256, 0, stream>>>(hH, WL16, bhhL, gL, P, 3 * HID, HID, 0, 0);
        gemm_wmma<_Float16><<<blocks, 256, 0, stream>>>(hH, WR16, bhhR, gR, P, 3 * HID, HID, 0, 0);
      }
    } else {
      const int blocks = ((P / 32) * ((3 * HID) / 32) + 7) / 8;
      gemm_wmma<float><<<blocks, 256, 0, stream>>>(hA, WhhL, bhhL, gL, P, 3 * HID, HID, 0, 0);
      gemm_wmma<float><<<blocks, 256, 0, stream>>>(hA, WhhR, bhhR, gR, P, 3 * HID, HID, 0, 0);
    }
    const int nt = P * HID;
    gru_gates<<<(nt + 255) / 256, 256, 0, stream>>>(hA, gL, gR, bihL, bihR, hB, P, 1 << l);
    float* t = hA; hA = hB; hB = t;
  }

  {  // logits = h_leaves * Wout^T + bout, stored transposed [leaf, b, v]
    if (pre) {
      const int nh = ROWS * HID;
      cvt_f32_f16<<<(nh + 255) / 256, 256, 0, stream>>>(hA, hH, nh);
      const int blocks = (ROWS / BM) * (VOCAB / BN);  // 16 * 250
      gemm_wmma_tiled<<<blocks, 256, 0, stream>>>(hH, WO16, bout, out, ROWS, VOCAB, HID, LEAVES, BATCH);
    } else {
      const int blocks = ((ROWS / 32) * (VOCAB / 32) + 7) / 8;
      gemm_wmma<float><<<blocks, 256, 0, stream>>>(hA, Wout, bout, out, ROWS, VOCAB, HID, LEAVES, BATCH);
    }
  }

  log_softmax_inplace<<<ROWS, 256, 0, stream>>>(out, VOCAB);
}
